// Conv_32083405701835
// MI455X (gfx1250) — compile-verified, matrix operand order checked by reading, not run on hardware
//
#include <hip/hip_runtime.h>

// ---------------------------------------------------------------------------
// Problem constants (match reference)
// ---------------------------------------------------------------------------
constexpr int cB = 2, cS = 2048, cH = 8, cHD = 64, cD = 512, cKS = 2048;
constexpr int cWIN = 64, cSUB = 129;          // window = [start, start+128]
constexpr int KEYSPAN = 160;                  // 16-aligned superset covering 16 sliding windows
constexpr int NKT = KEYSPAN / 16;             // 10 score tiles
constexpr int PS_STRIDE = 168;                // padded LDS row stride (bf16 elems)

typedef __attribute__((ext_vector_type(8)))  __bf16 v8bf;
typedef __attribute__((ext_vector_type(16))) __bf16 v16bf;
typedef __attribute__((ext_vector_type(8)))  float  v8f;

// Load a 16-element bf16 WMMA fragment for one lane.
// Layout (CDNA5 16-bit A/B striping): row fixed per lane; elements are
// K = kbase + 8*half + {0..7} and K = kbase + 8*half + 16 + {0..7}.
static __device__ __forceinline__ v16bf load_frag(const __bf16* row, int kbase, int half) {
  const __bf16* p = row + kbase + half * 8;
  v8bf lo = *(const v8bf*)(p);
  v8bf hi = *(const v8bf*)(p + 16);
  v16bf r;
#pragma unroll
  for (int i = 0; i < 8; ++i) { r[i] = lo[i]; r[i + 8] = hi[i]; }
  return r;
}

static __device__ __forceinline__ v8f wmma_bf16(v16bf a, v16bf b, v8f c) {
  // 8 args: (neg_a, A, neg_b, B, c_mod, C, reuse_a, reuse_b)
  return __builtin_amdgcn_wmma_f32_16x16x32_bf16(false, a, false, b, (short)0, c, false, false);
}

// ---------------------------------------------------------------------------
// Elementwise fp32 -> bf16 convert
// ---------------------------------------------------------------------------
__global__ __launch_bounds__(256) void cvt_bf16_kernel(const float* __restrict__ x,
                                                       __bf16* __restrict__ y, int n) {
  int i = blockIdx.x * 256 + threadIdx.x;
  if (i < n) y[i] = (__bf16)x[i];
}

// W[K][N] (fp32) -> Wt[N][K] (bf16): makes WMMA B-fragments contiguous per lane.
__global__ __launch_bounds__(256) void transpose_bf16_kernel(const float* __restrict__ W,
                                                             __bf16* __restrict__ Wt,
                                                             int K, int N) {
  int i = blockIdx.x * 256 + threadIdx.x;
  if (i < K * N) {
    int k = i / N, n = i - k * N;
    Wt[(size_t)n * K + k] = (__bf16)W[i];
  }
}

// ---------------------------------------------------------------------------
// Generic bf16 WMMA GEMM: out[M][N] = A[M][K] @ Wt[N][K]^T + bias (+relu/+resid)
// Block: 256 threads = 8 waves; each wave -> 32 rows x 64 cols (2x4 wmma tiles)
// so each B (weight) fragment feeds two WMMAs (halves redundant weight traffic;
// ~680 FLOP per L2 byte). No LDS: A rows and Wt rows give contiguous 16B
// per-lane fragment loads.
// MODE: 0 = f32 out, 1 = relu + bf16 out, 2 = +resid, f32 out
// ---------------------------------------------------------------------------
template <int MODE>
__global__ __launch_bounds__(256) void gemm_bf16_kernel(
    const __bf16* __restrict__ A, const __bf16* __restrict__ Wt,
    const float* __restrict__ bias, void* __restrict__ outv,
    const float* __restrict__ resid, int rstride, int M, int N, int K) {
  const int lane = threadIdx.x & 31;
  const int w    = threadIdx.x >> 5;
  const int n    = lane & 15;
  const int half = lane >> 4;
  const int r0   = blockIdx.y * 256 + w * 32;   // 32 rows per wave
  const int c0   = blockIdx.x * 64;

  v8f acc[2][4] = {};
  const __bf16* arow0 = A + (size_t)(r0 + n) * K;
  const __bf16* arow1 = A + (size_t)(r0 + 16 + n) * K;
  const __bf16* brow  = Wt + (size_t)(c0 + n) * K;

  for (int k0 = 0; k0 < K; k0 += 32) {
    const v16bf a0 = load_frag(arow0, k0, half);
    const v16bf a1 = load_frag(arow1, k0, half);
#pragma unroll
    for (int j = 0; j < 4; ++j) {
      const v16bf bb = load_frag(brow + (size_t)(j * 16) * K, k0, half);
      acc[0][j] = wmma_bf16(a0, bb, acc[0][j]);
      acc[1][j] = wmma_bf16(a1, bb, acc[1][j]);
    }
  }

#pragma unroll
  for (int j = 0; j < 4; ++j) {
    const int col = c0 + j * 16 + n;
    const float bc = bias[col];
#pragma unroll
    for (int rt = 0; rt < 2; ++rt) {
#pragma unroll
      for (int r = 0; r < 8; ++r) {
        const int row = r0 + rt * 16 + r + 8 * half;
        float v = acc[rt][j][r] + bc;
        if (MODE == 1) v = fmaxf(v, 0.0f);
        if (MODE == 2) v += resid[(size_t)row * rstride + col];
        if (MODE == 1) ((__bf16*)outv)[(size_t)row * N + col] = (__bf16)v;
        else           ((float*) outv)[(size_t)row * N + col] = v;
      }
    }
  }
}

// ---------------------------------------------------------------------------
// LayerNorm over k and q halves of kqv (bf16 out) + transposed bf16 copy of v.
// One block (256 thr) per token row. kqv row = [k(512) | q(512) | v(512)].
// v_t layout: [b][h][hd][S]  -> PV B-fragments contiguous in the key dim.
// ---------------------------------------------------------------------------
__global__ __launch_bounds__(256) void ln_split_kernel(
    const float* __restrict__ kqv, const float* __restrict__ gamma,
    const float* __restrict__ beta, __bf16* __restrict__ k_ln,
    __bf16* __restrict__ q_ln, __bf16* __restrict__ v_t) {
  __shared__ float sA[256], sB[256];
  const int row = blockIdx.x;
  const int b = row / cS, s = row - b * cS;
  const int t = threadIdx.x;
  const float* base = kqv + (size_t)row * (3 * cD);

#pragma unroll
  for (int part = 0; part < 2; ++part) {   // part 0 = k, part 1 = q
    const float* x = base + part * cD;
    const float v0 = x[t], v1 = x[t + 256];
    sA[t] = v0 + v1;
    sB[t] = v0 * v0 + v1 * v1;
    __syncthreads();
    for (int off = 128; off > 0; off >>= 1) {
      if (t < off) { sA[t] += sA[t + off]; sB[t] += sB[t + off]; }
      __syncthreads();
    }
    const float mu   = sA[0] * (1.0f / cD);
    const float var  = sB[0] * (1.0f / cD) - mu * mu;
    const float rstd = rsqrtf(var + 1e-3f);
    __bf16* dst = (part == 0) ? k_ln : q_ln;
    dst[(size_t)row * cD + t]       = (__bf16)((v0 - mu) * rstd * gamma[t]       + beta[t]);
    dst[(size_t)row * cD + t + 256] = (__bf16)((v1 - mu) * rstd * gamma[t + 256] + beta[t + 256]);
    __syncthreads();
  }

#pragma unroll
  for (int d = t; d < cD; d += 256) {
    v_t[((size_t)(b * cH + (d >> 6)) * cHD + (d & 63)) * cS + s] = (__bf16)base[2 * cD + d];
  }
}

// ---------------------------------------------------------------------------
// Sliding-window attention, one wave per (batch, head, 16-query block).
// QK^T and PV via v_wmma_f32_16x16x32_bf16; masked shuffle softmax;
// P re-layout (D-form -> A-form) through padded LDS.
// ---------------------------------------------------------------------------
__global__ __launch_bounds__(32) void attn_kernel(
    const __bf16* __restrict__ q_ln, const __bf16* __restrict__ k_ln,
    const __bf16* __restrict__ v_t, __bf16* __restrict__ ctx) {
  __shared__ __bf16 Ps[16 * PS_STRIDE];

  const int bid  = blockIdx.x;
  const int sblk = bid & (cS / 16 - 1);
  const int h    = (bid >> 7) & (cH - 1);
  const int bb   = bid >> 10;
  const int s0   = sblk << 4;

  int kb = s0 - cWIN;
  if (kb < 0) kb = 0;
  if (kb > cS - KEYSPAN) kb = cS - KEYSPAN;   // 16-aligned: s0 and cS-KEYSPAN both are

  const int lane = threadIdx.x;
  const int n    = lane & 15;
  const int half = lane >> 4;

  // Q fragments (16 queries x 64 head-dim -> two K-steps of 32)
  const __bf16* qrow = q_ln + ((size_t)(bb * cS + s0 + n)) * cD + h * cHD;
  const v16bf a0 = load_frag(qrow, 0, half);
  const v16bf a1 = load_frag(qrow, 32, half);

  // Scores: 10 tiles of 16 keys
  v8f sc[NKT];
#pragma unroll
  for (int t = 0; t < NKT; ++t) {
    const int key = kb + t * 16 + n;
    const __bf16* krow = k_ln + ((size_t)(bb * cS + key)) * cD + h * cHD;
    v8f c = {};
    c = wmma_bf16(a0, load_frag(krow, 0, half), c);
    c = wmma_bf16(a1, load_frag(krow, 32, half), c);
    sc[t] = c;
  }

  // Masked softmax (row r lives in one 16-lane half-group)
#pragma unroll
  for (int r = 0; r < 8; ++r) {
    const int sq = s0 + r + 8 * half;
    int start = sq - cWIN;
    if (start < 0) start = 0;
    if (start > cS - cSUB) start = cS - cSUB;
    float m = -1e30f;
#pragma unroll
    for (int t = 0; t < NKT; ++t) {
      const int key = kb + t * 16 + n;
      float v = sc[t][r] * 0.125f;                       // 1/sqrt(64)
      const bool valid = (key >= start) && (key < start + cSUB);
      v = valid ? v : -1e30f;
      sc[t][r] = v;
      m = fmaxf(m, v);
    }
#pragma unroll
    for (int off = 1; off < 16; off <<= 1) m = fmaxf(m, __shfl_xor(m, off, 32));
    float sum = 0.0f;
#pragma unroll
    for (int t = 0; t < NKT; ++t) {
      const float e = __expf(sc[t][r] - m);
      sc[t][r] = e;
      sum += e;
    }
#pragma unroll
    for (int off = 1; off < 16; off <<= 1) sum += __shfl_xor(sum, off, 32);
    const float inv = 1.0f / sum;
#pragma unroll
    for (int t = 0; t < NKT; ++t)
      Ps[(r + 8 * half) * PS_STRIDE + t * 16 + n] = (__bf16)(sc[t][r] * inv);
  }
  __syncthreads();

  // ctx = P[16x160] @ Vwin[160x64]  (5 K-steps of 32, 4 output tiles)
  v8f acc[4] = {};
  const __bf16* prow = &Ps[n * PS_STRIDE];
  const size_t vbase = (size_t)(bb * cH + h) * cHD;
#pragma unroll
  for (int step = 0; step < 5; ++step) {
    const v16bf ap = load_frag(prow, step * 32, half);
#pragma unroll
    for (int j = 0; j < 4; ++j) {
      const __bf16* vrow = v_t + (vbase + j * 16 + n) * cS + kb;
      acc[j] = wmma_bf16(ap, load_frag(vrow, step * 32, half), acc[j]);
    }
  }

#pragma unroll
  for (int j = 0; j < 4; ++j)
#pragma unroll
    for (int r = 0; r < 8; ++r)
      ctx[((size_t)(bb * cS + s0 + r + 8 * half)) * cD + h * cHD + j * 16 + n] =
          (__bf16)acc[j][r];
}

// ---------------------------------------------------------------------------
// Host launcher
// ---------------------------------------------------------------------------
extern "C" void kernel_launch(void* const* d_in, const int* in_sizes, int n_in,
                              void* d_out, int out_size, void* d_ws, size_t ws_size,
                              hipStream_t stream) {
  const float* values  = (const float*)d_in[0];
  const float* W_kqv   = (const float*)d_in[1];
  const float* b_kqv   = (const float*)d_in[2];
  const float* g       = (const float*)d_in[3];
  const float* be      = (const float*)d_in[4];
  const float* W_ker   = (const float*)d_in[5];
  const float* b_ker   = (const float*)d_in[6];
  const float* W_proj  = (const float*)d_in[7];
  const float* b_proj  = (const float*)d_in[8];
  float* out = (float*)d_out;

  const int BS = cB * cS;  // 4096 token rows
  size_t off = 0;
  auto alloc = [&](size_t bytes) -> char* {
    char* p = (char*)d_ws + off;
    off += (bytes + 255) & ~(size_t)255;
    return p;
  };
  __bf16* vals_bf = (__bf16*)alloc((size_t)BS * cD * 2);
  __bf16* Wkqv_t  = (__bf16*)alloc((size_t)cD * 3 * cD * 2);
  __bf16* Wker_t  = (__bf16*)alloc((size_t)cD * cKS * 2);
  __bf16* Wproj_t = (__bf16*)alloc((size_t)cKS * cD * 2);
  float*  kqv     = (float*) alloc((size_t)BS * 3 * cD * 4);
  __bf16* k_ln    = (__bf16*)alloc((size_t)BS * cD * 2);
  __bf16* q_ln    = (__bf16*)alloc((size_t)BS * cD * 2);
  __bf16* v_t     = (__bf16*)alloc((size_t)cB * cH * cHD * cS * 2);
  __bf16* ctx     = (__bf16*)alloc((size_t)BS * cD * 2);
  __bf16* hidden  = (__bf16*)alloc((size_t)BS * cKS * 2);

  // Precision conversion / weight transposes (tiny, one-time per launch)
  {
    int n = BS * cD;
    cvt_bf16_kernel<<<(n + 255) / 256, 256, 0, stream>>>(values, vals_bf, n);
  }
  transpose_bf16_kernel<<<(cD * 3 * cD + 255) / 256, 256, 0, stream>>>(W_kqv, Wkqv_t, cD, 3 * cD);
  transpose_bf16_kernel<<<(cD * cKS + 255) / 256, 256, 0, stream>>>(W_ker, Wker_t, cD, cKS);
  transpose_bf16_kernel<<<(cKS * cD + 255) / 256, 256, 0, stream>>>(W_proj, Wproj_t, cKS, cD);

  // G1: kqv = values @ W_kqv + b_kqv   (f32 out, needed for LN + residual v)
  gemm_bf16_kernel<0><<<dim3(3 * cD / 64, BS / 256), 256, 0, stream>>>(
      vals_bf, Wkqv_t, b_kqv, kqv, nullptr, 0, BS, 3 * cD, cD);

  // LN(k), LN(q) -> bf16; v -> transposed bf16
  ln_split_kernel<<<BS, 256, 0, stream>>>(kqv, g, be, k_ln, q_ln, v_t);

  // Sliding-window attention -> ctx (bf16)
  attn_kernel<<<cB * cH * (cS / 16), 32, 0, stream>>>(q_ln, k_ln, v_t, ctx);

  // G2: hidden = relu(ctx @ W_kernel + b_kernel)  (bf16 out)
  gemm_bf16_kernel<1><<<dim3(cKS / 64, BS / 256), 256, 0, stream>>>(
      ctx, Wker_t, b_ker, hidden, nullptr, 0, BS, cKS, cD);

  // G3: out = hidden @ W_proj + b_proj + v   (v = kqv[:, 1024:1536], f32 out)
  gemm_bf16_kernel<2><<<dim3(cD / 64, BS / 256), 256, 0, stream>>>(
      hidden, Wproj_t, b_proj, out, kqv + 2 * cD, 3 * cD, BS, cD, cKS);
}
